// hierarchically_modular_cnn_35321811042826
// MI455X (gfx1250) — matrix operand order repaired and checked by reading, not compiled, not run on hardware
//
#include <hip/hip_runtime.h>
#include <hip/hip_bf16.h>
#include <stdint.h>

typedef __attribute__((ext_vector_type(16))) _Float16 v16h;
typedef __attribute__((ext_vector_type(8)))  _Float16 v8h;
typedef __attribute__((ext_vector_type(8)))  float    v8f;
typedef __attribute__((ext_vector_type(4)))  unsigned u32x4;
typedef __attribute__((ext_vector_type(8)))  unsigned u32x8;

#define NB   8192
#define NE   4
#define IMGS 8      // images per workgroup tile

// ---------------- routing: argmax / top-2 index extraction ----------------
__global__ void route_kernel(const float* __restrict__ emb_img,
                             const float* __restrict__ emb_mlp0,
                             const float* __restrict__ emb_mlp1,
                             const float* __restrict__ emb_out,
                             int* __restrict__ route) {
  int tid = threadIdx.x;
  if (tid < 4) {                        // per-expert channel argmax (ST softmax forward)
    float best = emb_img[tid]; int bi = 0;
    for (int c = 1; c < 4; ++c) { float v = emb_img[c * 4 + tid]; if (v > best) { best = v; bi = c; } }
    route[tid] = bi;
  } else if (tid >= 8 && tid < 16) {    // layer0 top-2 over D=32
    int m = tid - 8;
    float m1 = -1e30f, m2 = -1e30f; int i1 = 0, i2 = 0;
    for (int d = 0; d < 32; ++d) {
      float v = emb_mlp0[d * 8 + m];
      if (v > m1) { m2 = m1; i2 = i1; m1 = v; i1 = d; }
      else if (v > m2) { m2 = v; i2 = d; }
    }
    route[4 + m] = i1; route[12 + m] = i2;
  } else if (tid >= 16 && tid < 20) {   // layer1 top-2 over D=8
    int m = tid - 16;
    float m1 = -1e30f, m2 = -1e30f; int i1 = 0, i2 = 0;
    for (int d = 0; d < 8; ++d) {
      float v = emb_mlp1[d * 4 + m];
      if (v > m1) { m2 = m1; i2 = i1; m1 = v; i1 = d; }
      else if (v > m2) { m2 = v; i2 = d; }
    }
    route[20 + m] = i1; route[24 + m] = i2;
  } else if (tid == 20) {               // head top-2 over D=4
    float m1 = -1e30f, m2 = -1e30f; int i1 = 0, i2 = 0;
    for (int d = 0; d < 4; ++d) {
      float v = emb_out[d];
      if (v > m1) { m2 = m1; i2 = i1; m1 = v; i1 = d; }
      else if (v > m2) { m2 = v; i2 = d; }
    }
    route[28] = i1; route[29] = i2;
  }
}

// ------- pack conv2/fc weights into per-lane WMMA B-fragment order (f16) -------
// B (32x16 f16) layout: lane column n = lane&15; element e -> K = 8*(lane>>4) + e (e<8), 16 + 8*(lane>>4) + (e-8) (e>=8).
// Stored as [expert][kstep][lane][16 f16] so a lane's fragment is one 32B load.
__global__ __launch_bounds__(256) void pack_kernel(const float* __restrict__ conv2_w,
                                                   const float* __restrict__ fc_w,
                                                   _Float16* __restrict__ w2frag,
                                                   _Float16* __restrict__ fcfrag) {
  int idx = blockIdx.x * 256 + threadIdx.x;
  if (idx < NE * 5 * 512) {             // conv2: K=150 padded to 160 (5 k-steps), N=16
    int e = idx / 2560, r = idx - e * 2560;
    int ks = r / 512,  lr = r - ks * 512;
    int lane = lr >> 4, el = lr & 15;
    int k = ks * 32 + ((el < 8) ? ((lane >> 4) * 8 + el) : (16 + (lane >> 4) * 8 + (el - 8)));
    int n = lane & 15;
    float v = 0.f;
    if (k < 150) v = conv2_w[(e * 16 + n) * 150 + k];   // [e][n][c*25+dy*5+dx]
    w2frag[idx] = (_Float16)v;
  } else {
    int j = idx - NE * 5 * 512;
    if (j < NE * 8 * 512) {             // fc: K=256 (8 k-steps), N=8 padded to 16
      int e = j / 4096, r = j - e * 4096;
      int ks = r / 512,  lr = r - ks * 512;
      int lane = lr >> 4, el = lr & 15;
      int k = ks * 32 + ((el < 8) ? ((lane >> 4) * 8 + el) : (16 + (lane >> 4) * 8 + (el - 8)));
      int n = lane & 15;
      fcfrag[j] = (_Float16)((n < 8) ? fc_w[(e * 8 + n) * 256 + k] : 0.f);
    }
  }
}

// ---------------- fused CNN expert pipeline: one WG per (expert, 8 images) ----------------
__global__ __launch_bounds__(256) void cnn_kernel(
    const float* __restrict__ x, const float* __restrict__ conv1_w,
    const float* __restrict__ conv1_b, const float* __restrict__ conv2_b,
    const float* __restrict__ fc_b, const int* __restrict__ route,
    const _Float16* __restrict__ w2frag, const _Float16* __restrict__ fcfrag,
    float* __restrict__ h0) {
  __shared__ alignas(16) float    s_img[IMGS * 784];        // 25.1 KB selected-channel images
  __shared__ float    s_w1[150];
  __shared__ float    s_b1[6];
  __shared__ alignas(16) _Float16 s_pool1[IMGS * 6 * 144];  // 13.8 KB pooled conv1 (f16)
  __shared__ alignas(16) _Float16 s_c2[IMGS * 16 * 64];     // 16.4 KB conv2 out (relu'd f16)
  __shared__ alignas(16) _Float16 s_fcin[IMGS * 256];       //  4.1 KB pooled conv2 (fc input)

  const int tid = threadIdx.x;
  const int e = blockIdx.x & (NE - 1);
  const int img_base = (blockIdx.x >> 2) * IMGS;
  const int sel = route[e];
  const int lane = tid & 31;
  const int wv = tid >> 5;      // wave id 0..7
  const int g = lane >> 4;      // lane half
  const int ln = lane & 15;     // A row (pixel) / B,D column (channel)

  for (int i = tid; i < 150; i += 256) s_w1[i] = conv1_w[e * 150 + i];
  if (tid < 6) s_b1[tid] = conv1_b[e * 6 + tid];

  // ---- Tensor Data Mover: DMA the 8-image tile (784 x 8 rows, row stride 3136
  // f32 elements) straight into LDS.  D# per CDNA5 ISA.
  if (wv == 0) {
    const unsigned lds_base = (unsigned)(uintptr_t)(void*)s_img;     // low 32b = LDS offset
    const unsigned long long gaddr =
        (unsigned long long)(uintptr_t)&x[(img_base * 4 + sel) * 784];
    u32x4 g0;
    g0[0] = 1u;                                          // count=1 (valid user D#)
    g0[1] = lds_base;                                    // lds_addr
    g0[2] = (unsigned)(gaddr & 0xffffffffu);             // global_addr[31:0]
    g0[3] = (unsigned)((gaddr >> 32) & 0x01ffffffu)      // global_addr[56:32]
            | (2u << 30);                                // type = 2 ("image")
    u32x8 g1;
    g1[0] = 2u << 16;                                    // data_size = 4 bytes
    g1[1] = (784u & 0xffffu) << 16;                      // tensor_dim0[15:0] @ bit48
    g1[2] = (784u >> 16) | (8u << 16);                   // tensor_dim0[31:16] | tensor_dim1[15:0]
    g1[3] = (784u << 16);                                // tensor_dim1[31:16] | tile_dim0 = 784
    g1[4] = 8u;                                          // tile_dim1 = 8, tile_dim2 = 0
    g1[5] = 3136u;                                       // tensor_dim0_stride[31:0] = 4*784
    g1[6] = 0u;                                          // stride hi | tensor_dim1_stride lo
    g1[7] = 0u;
    asm volatile("tensor_load_to_lds %0, %1" :: "s"(g0), "s"(g1) : "memory");
    __builtin_amdgcn_s_wait_tensorcnt(0);
  }
  if (img_base + IMGS < NB) {   // CDNA5 global_prefetch_b8: pull next tile toward L2
    __builtin_prefetch(&x[((img_base + IMGS + (tid & 7)) * 4 + sel) * 784 + (tid >> 3) * 24], 0, 1);
  }
  __syncthreads();

  // conv1 (K=25, too shallow for WMMA reformat) + relu + 2x2 maxpool, f32 VALU -> f16 LDS
  for (int p = tid; p < IMGS * 6 * 144; p += 256) {
    int im = p / 864, r = p - im * 864;
    int ch = r / 144, q = r - ch * 144;
    int py = q / 12,  px = q - py * 12;
    const float* wp = &s_w1[ch * 25];
    const float bias = s_b1[ch];
    float mx = 0.f;
    #pragma unroll
    for (int sy = 0; sy < 2; ++sy)
      #pragma unroll
      for (int sx = 0; sx < 2; ++sx) {
        const float* ip = &s_img[im * 784 + (2 * py + sy) * 28 + (2 * px + sx)];
        float acc = bias;
        #pragma unroll
        for (int dy = 0; dy < 5; ++dy)
          #pragma unroll
          for (int dx = 0; dx < 5; ++dx)
            acc = fmaf(ip[dy * 28 + dx], wp[dy * 5 + dx], acc);
        mx = fmaxf(mx, fmaxf(acc, 0.f));
      }
    s_pool1[p] = (_Float16)mx;
  }
  __syncthreads();

  // conv2 as implicit GEMM: M = 8 img * 64 px = 512 rows (32 tiles), N = 16, K = 150->160
  v16h bf[5];
  {
    const v16h* bp = (const v16h*)(w2frag + e * 5 * 512);
    #pragma unroll
    for (int ks = 0; ks < 5; ++ks) bf[ks] = bp[ks * 32 + lane];
  }
  const float c2bias = conv2_b[e * 16 + ln];

  for (int t = wv * 4; t < wv * 4 + 4; ++t) {   // 4 M-tiles per wave (wave-uniform)
    const int im = t >> 2;
    const int pbase = (t & 3) * 16;
    const int pixA = pbase + ln;                // conv2 output pixel for this A row
    const int oy = pixA >> 3, ox = pixA & 7;
    v8f acc = {};
    #pragma unroll
    for (int ks = 0; ks < 5; ++ks) {
      v16h a;                                   // 16-bit A 16x32 layout (ISA 7.12.2)
      #pragma unroll
      for (int el = 0; el < 16; ++el) {
        const int k = ks * 32 + ((el < 8) ? (g * 8 + el) : (16 + g * 8 + (el - 8)));
        _Float16 v = (_Float16)0.f;
        if (k < 150) {
          const int c = k / 25, rr = k - c * 25;
          const int dy = rr / 5, dx = rr - dy * 5;
          v = s_pool1[(im * 6 + c) * 144 + (oy + dy) * 12 + (ox + dx)];
        }
        a[el] = v;
      }
      acc = __builtin_amdgcn_wmma_f32_16x16x32_f16(false, a, false, bf[ks],
                                                   (short)0, acc, false, false);
    }
    #pragma unroll
    for (int r2 = 0; r2 < 8; ++r2) {            // D: VGPR r holds M = r + 8*half
      const int pix = pbase + r2 + 8 * g;
      s_c2[(im * 16 + ln) * 64 + pix] = (_Float16)fmaxf(acc[r2] + c2bias, 0.f);
    }
  }
  __syncthreads();

  // 2x2 maxpool conv2 -> fc input, reshape order c*16 + py*4 + px
  for (int p = tid; p < IMGS * 256; p += 256) {
    const int im = p >> 8, q = p & 255;
    const int ch = q >> 4, q2 = q & 15;
    const int py = q2 >> 2, px = q2 & 3;
    const _Float16* cp = &s_c2[(im * 16 + ch) * 64 + (2 * py) * 8 + 2 * px];
    s_fcin[p] = (_Float16)fmaxf(fmaxf((float)cp[0], (float)cp[1]),
                                fmaxf((float)cp[8], (float)cp[9]));
  }
  __syncthreads();

  // FC 256->8 via WMMA: M = 8 imgs (padded to 16), K = 256 (8 k-steps), N = 8 pad 16.
  // A half-fragments are K-contiguous runs of 8 -> two 16B ds loads per fragment.
  if (wv == 0) {   // wave-uniform branch: EXEC all-1s inside, WMMA legal
    const v16h* bp = (const v16h*)(fcfrag + e * 8 * 512);
    const v8h* rp = (const v8h*)s_fcin + ln * 32;   // 32 v8h per 256-elem row
    const v16h zero = {};
    v8f accf = {};
    #pragma unroll
    for (int ks = 0; ks < 8; ++ks) {
      const v8h lo = rp[ks * 4 + g];                // K = ks*32 + 8g .. +7
      const v8h hi = rp[ks * 4 + g + 2];            // K = ks*32 + 16 + 8g .. +7
      v16h a = __builtin_shufflevector(lo, hi, 0, 1, 2, 3, 4, 5, 6, 7,
                                               8, 9, 10, 11, 12, 13, 14, 15);
      a = (ln < IMGS) ? a : zero;                   // pad rows 8..15 of the M tile
      accf = __builtin_amdgcn_wmma_f32_16x16x32_f16(false, a, false, bp[ks * 32 + lane],
                                                    (short)0, accf, false, false);
    }
    const float fb = (ln < 8) ? fc_b[e * 8 + ln] : 0.f;
    #pragma unroll
    for (int r2 = 0; r2 < 8; ++r2) {
      const int M = r2 + 8 * g;                 // image row
      if (M < IMGS && ln < 8)
        h0[(img_base + M) * 32 + e * 8 + ln] = accf[r2] + fb;   // [B, E*8] concat order
    }
  }
}

// ---------------- fused routed-MLP layers + head ----------------
__global__ __launch_bounds__(256) void mlp_head_kernel(
    const float* __restrict__ h0, const int* __restrict__ route,
    const float* __restrict__ w10, const float* __restrict__ b10,
    const float* __restrict__ w20, const float* __restrict__ b20,
    const float* __restrict__ w11, const float* __restrict__ b11,
    const float* __restrict__ w21, const float* __restrict__ b21,
    float* __restrict__ out) {
  __shared__ float sp[3084];
  __shared__ float s_h1[256 * 8];
  __shared__ float s_h2[256 * 4];
  const int tid = threadIdx.x;
  for (int i = tid; i < 1024; i += 256) sp[i] = w10[i];
  for (int i = tid; i < 512;  i += 256) sp[1024 + i] = b10[i];
  for (int i = tid; i < 512;  i += 256) sp[1536 + i] = w20[i];
  if (tid < 8) sp[2048 + tid] = b20[tid];
  for (int i = tid; i < 512;  i += 256) sp[2056 + i] = w11[i];
  for (int i = tid; i < 256;  i += 256) sp[2568 + i] = b11[i];
  for (int i = tid; i < 256;  i += 256) sp[2824 + i] = w21[i];
  if (tid < 4) sp[3080 + tid] = b21[tid];
  __syncthreads();

  const int b = blockIdx.x * 256 + tid;
  const float* hr = &h0[b * 32];
  for (int m = 0; m < 8; ++m) {
    const float x0 = hr[route[4 + m]], x1 = hr[route[12 + m]];
    const float* w1p = &sp[m * 128];
    const float* b1p = &sp[1024 + m * 64];
    const float* w2p = &sp[1536 + m * 64];
    float acc = 0.f;
    #pragma unroll 8
    for (int hh = 0; hh < 64; ++hh) {
      float t = fmaf(x0, w1p[hh * 2], fmaf(x1, w1p[hh * 2 + 1], b1p[hh]));
      acc = fmaf(fmaxf(t, 0.f), w2p[hh], acc);
    }
    s_h1[tid * 8 + m] = acc + sp[2048 + m];
  }
  for (int m = 0; m < 4; ++m) {
    const float x0 = s_h1[tid * 8 + route[20 + m]], x1 = s_h1[tid * 8 + route[24 + m]];
    const float* w1p = &sp[2056 + m * 128];
    const float* b1p = &sp[2568 + m * 64];
    const float* w2p = &sp[2824 + m * 64];
    float acc = 0.f;
    #pragma unroll 8
    for (int hh = 0; hh < 64; ++hh) {
      float t = fmaf(x0, w1p[hh * 2], fmaf(x1, w1p[hh * 2 + 1], b1p[hh]));
      acc = fmaf(fmaxf(t, 0.f), w2p[hh], acc);
    }
    s_h2[tid * 4 + m] = acc + sp[3080 + m];
  }
  out[b * 2 + 0] = 1.f / (1.f + __expf(-s_h2[tid * 4 + route[28]]));
  out[b * 2 + 1] = 1.f / (1.f + __expf(-s_h2[tid * 4 + route[29]]));
}

extern "C" void kernel_launch(void* const* d_in, const int* in_sizes, int n_in,
                              void* d_out, int out_size, void* d_ws, size_t ws_size,
                              hipStream_t stream) {
  const float* x        = (const float*)d_in[0];
  const float* emb_img  = (const float*)d_in[1];
  const float* conv1_w  = (const float*)d_in[2];
  const float* conv1_b  = (const float*)d_in[3];
  const float* conv2_w  = (const float*)d_in[4];
  const float* conv2_b  = (const float*)d_in[5];
  const float* fc_w     = (const float*)d_in[6];
  const float* fc_b     = (const float*)d_in[7];
  const float* emb_mlp0 = (const float*)d_in[8];
  const float* w10      = (const float*)d_in[9];
  const float* b10      = (const float*)d_in[10];
  const float* w20      = (const float*)d_in[11];
  const float* b20      = (const float*)d_in[12];
  const float* emb_mlp1 = (const float*)d_in[13];
  const float* w11      = (const float*)d_in[14];
  const float* b11      = (const float*)d_in[15];
  const float* w21      = (const float*)d_in[16];
  const float* b21      = (const float*)d_in[17];
  const float* emb_out  = (const float*)d_in[18];

  char* ws = (char*)d_ws;                       // needs ~1.1 MB
  int*       route  = (int*)ws;                 // 32 ints
  _Float16*  w2frag = (_Float16*)(ws + 256);    // 4*5*512 f16 = 20480 B
  _Float16*  fcfrag = (_Float16*)(ws + 20736);  // 4*8*512 f16 = 32768 B
  float*     h0     = (float*)(ws + 53504);     // 8192*32 f32 = 1 MB

  route_kernel<<<1, 32, 0, stream>>>(emb_img, emb_mlp0, emb_mlp1, emb_out, route);
  pack_kernel<<<104, 256, 0, stream>>>(conv2_w, fc_w, w2frag, fcfrag);
  cnn_kernel<<<(NB / IMGS) * NE, 256, 0, stream>>>(x, conv1_w, conv1_b, conv2_b, fc_b,
                                                   route, w2frag, fcfrag, h0);
  mlp_head_kernel<<<NB / 256, 256, 0, stream>>>(h0, route, w10, b10, w20, b20,
                                                w11, b11, w21, b21, (float*)d_out);
}
